// cnnVoting_22960895164472
// MI455X (gfx1250) — compile-verified, matrix operand order checked by reading, not run on hardware
//
#include <hip/hip_runtime.h>
#include <hip/hip_bf16.h>

// ---------------------------------------------------------------------------
// CDNA5 (gfx1250) fused CNN inference: 4x [conv5x5+BN+ReLU+pool2x2] -> FC -> FC
// Convs are implicit GEMM on v_wmma_f32_16x16x32_f16 (wave32), f16 activations
// in NHWC, BN folded into f16 weights + f32 bias, pooling fused via LDS.
// Halo staging uses GLOBAL_LOAD_ASYNC_TO_LDS_B128 (ASYNCcnt) when available.
// ---------------------------------------------------------------------------

typedef _Float16 v8h  __attribute__((ext_vector_type(8)));
typedef _Float16 v16h __attribute__((ext_vector_type(16)));
typedef float    v8f  __attribute__((ext_vector_type(8)));
typedef float    v4f  __attribute__((ext_vector_type(4)));

#if defined(__gfx1250__) &&                                                    \
    __has_builtin(__builtin_amdgcn_global_load_async_to_lds_b128) &&           \
    __has_builtin(__builtin_amdgcn_s_wait_asynccnt)
#define HAS_ASYNC_LDS 1
typedef int v4i __attribute__((vector_size(4 * sizeof(int))));
typedef __attribute__((address_space(1))) v4i* gas_p;   // global int4*
typedef __attribute__((address_space(3))) v4i* las_p;   // LDS int4*
#else
#define HAS_ASYNC_LDS 0
#endif

#define DEVINL __device__ __forceinline__

DEVINL v16h concat8(v8h a0, v8h a1) {
    return __builtin_shufflevector(a0, a1, 0, 1, 2, 3, 4, 5, 6, 7,
                                   8, 9, 10, 11, 12, 13, 14, 15);
}

// A-fragment: lane holds M = cbase + (lane&15); K offsets {kbase..kbase+7} and
// {kbase+16..kbase+23} with kbase = (lane<16 ? 0 : 8).  Weights are packed
// row-major [Cout][KP] f16 so both chunks are contiguous 16B loads.
DEVINL v16h load_a_frag(const _Float16* __restrict__ wr, int kbase) {
    v8h a0 = *(const v8h*)(wr + kbase);
    v8h a1 = *(const v8h*)(wr + kbase + 16);
    return concat8(a0, a1);
}

// B-fragment gathered from the LDS halo tile laid out [ROWS][COLS][CIN] f16.
// K index = tap*CIN + c, tap = dy*5+dx.
//  - CIN>=16: each 8-element group is a contiguous channel run in one tap
//             -> one ds_load_b128 per group.
//  - CIN==1 (layer 1): all offsets compile-time; the only lane-varying term is
//             hi (kbase 0 vs 8) -> v_cndmask-selected immediate offsets, no
//             branches.  K in [25,32) reads in-bounds junk rows that are
//             annihilated by zero-padded weights (A=0), so no guard needed.
template <int CIN, int COLS>
DEVINL v16h load_b_frag(const _Float16* __restrict__ sm, int rt, int lcol0,
                        int kk, int kbase, int hi) {
    if constexpr (CIN >= 16) {
        constexpr int KTOT = 25 * CIN;
        v8h g[2];
#pragma unroll
        for (int gi = 0; gi < 2; ++gi) {
            const int K0 = kk + kbase + gi * 16;
            if (K0 < KTOT) {
                const int tap = K0 / CIN;      // CIN is a power of two
                const int c0  = K0 % CIN;
                const int dy = tap / 5, dx = tap % 5;
                const int idx = ((rt + dy) * COLS + (lcol0 + dx)) * CIN + c0;
                g[gi] = *(const v8h*)(sm + idx);
            } else {
                g[gi] = (v8h){(_Float16)0, (_Float16)0, (_Float16)0, (_Float16)0,
                              (_Float16)0, (_Float16)0, (_Float16)0, (_Float16)0};
            }
        }
        return concat8(g[0], g[1]);
    } else {
        const _Float16* p = sm + rt * COLS + lcol0;
        v16h b;
#pragma unroll
        for (int e = 0; e < 16; ++e) {
            const int k0 = (e < 8) ? e : (8 + e);   // kbase==0 K value
            const int k1 = k0 + 8;                  // kbase==8 K value
            const int o0 = (k0 / 5) * COLS + (k0 % 5);
            const int o1 = (k1 / 5) * COLS + (k1 % 5);
            b[e] = p[hi ? o1 : o0];
        }
        return b;
    }
}

// ---------------------------------------------------------------------------
// Fused Conv(5x5,pad2) + folded-BN + ReLU + MaxPool(2x2)   (NHWC f16 -> f16)
// Block = 128 threads = 4 waves. Block covers 2 conv rows x (32*SP) conv cols
// of one image; wave (w) -> cout tile (w%TCO) and spatial subtile (w/TCO).
// ---------------------------------------------------------------------------
template <int CIN, int COUT, int HIN, int WIN>
__global__ void __launch_bounds__(128)
conv_bnrp_kernel(const _Float16* __restrict__ in,
                 const _Float16* __restrict__ wp,
                 const float* __restrict__ bias,
                 _Float16* __restrict__ out) {
    constexpr int TCO  = COUT / 16;       // cout tiles per block (1,2,4,2)
    constexpr int SP   = 4 / TCO;         // spatial subtiles per block
    constexpr int HP   = HIN / 2;
    constexpr int WPD  = WIN / 2;
    constexpr int KTOT = 25 * CIN;
    constexpr int KP   = ((KTOT + 31) / 32) * 32;
    constexpr int COLS = 32 * SP + 4;     // halo'd col extent
    constexpr int ROWS = (CIN == 1) ? 8 : 6;  // extra rows absorb padded K taps
    constexpr int XB   = (WPD + 16 * SP - 1) / (16 * SP);

    __shared__ alignas(16) _Float16 smem[ROWS * COLS * CIN];
    __shared__ alignas(16) _Float16 pool[4][2][32][16]; // [wave][row][col][co]

    const int tid  = threadIdx.x;
    const int wave = tid >> 5;
    const int lane = tid & 31;

    int bid = blockIdx.x;
    const int xb = bid % XB; bid /= XB;
    const int py = bid % HP;
    const int n  = bid / HP;
    const int r0  = 2 * py;
    const int sc0 = xb * 32 * SP;

    // ---- stage input halo tile (zero-padded) into LDS ----
    if constexpr (CIN >= 16) {
        constexpr int NV = ROWS * COLS * CIN / 8;   // 16B packets
        for (int e8 = tid; e8 < NV; e8 += 128) {
            const int e   = e8 * 8;
            const int c   = e % CIN;
            const int t2  = e / CIN;
            const int col = t2 % COLS;
            const int row = t2 / COLS;
            const int gr = r0 - 2 + row;
            const int gc = sc0 - 2 + col;
            if (gr >= 0 && gr < HIN && gc >= 0 && gc < WIN) {
                const _Float16* src =
                    &in[((size_t)(n * HIN + gr) * WIN + gc) * CIN + c];
#if HAS_ASYNC_LDS
                // async Global->LDS DMA, tracked by ASYNCcnt
                __builtin_amdgcn_global_load_async_to_lds_b128(
                    (gas_p)src, (las_p)&smem[e], 0, 0);
#else
                *(v8h*)&smem[e] = *(const v8h*)src;
#endif
            } else {
                *(v8h*)&smem[e] =
                    (v8h){(_Float16)0, (_Float16)0, (_Float16)0, (_Float16)0,
                          (_Float16)0, (_Float16)0, (_Float16)0, (_Float16)0};
            }
        }
#if HAS_ASYNC_LDS
        __builtin_amdgcn_s_wait_asynccnt(0);
#endif
    } else {
        for (int e = tid; e < ROWS * COLS; e += 128) {
            const int col = e % COLS;
            const int row = e / COLS;
            const int gr = r0 - 2 + row;
            const int gc = sc0 - 2 + col;
            _Float16 v = (_Float16)0;
            if (gr >= 0 && gr < HIN && gc >= 0 && gc < WIN)
                v = in[(size_t)(n * HIN + gr) * WIN + gc];
            smem[e] = v;
        }
    }
    __syncthreads();

    const int ctile = wave % TCO;
    const int sub   = wave / TCO;
    const int cbase = ctile * 16;
    const int nlo   = lane & 15;
    const int hi    = (lane >> 4) & 1;
    const int kbase = hi * 8;
    const int wc0   = sub * 32;  // wave's conv-col base, relative to sc0

    const v8f vzero = {0.f, 0.f, 0.f, 0.f, 0.f, 0.f, 0.f, 0.f};
    v8f acc[4];
#pragma unroll
    for (int t = 0; t < 4; ++t) acc[t] = vzero;

    const _Float16* wrow = wp + (size_t)(cbase + nlo) * KP;
    for (int kk = 0; kk < KP; kk += 32) {
        __builtin_prefetch(wrow + kk + 64, 0, 0);   // global_prefetch on weights
        const v16h a = load_a_frag(wrow + kk, kbase);
#pragma unroll
        for (int t = 0; t < 4; ++t) {
            const int rt = t >> 1, ct = t & 1;
            const int lcol0 = wc0 + ct * 16 + nlo;
            const v16h b =
                load_b_frag<CIN, COLS>(smem, rt, lcol0, kk, kbase, hi);
            acc[t] = __builtin_amdgcn_wmma_f32_16x16x32_f16(
                false, a, false, b, (short)0, acc[t], false, false);
        }
    }

    // ---- spill raw conv accumulators (f16) for 2x2 pooling ----
    // lane's 8 acc elements are contiguous in cout -> one b128 store per tile.
    // (per-channel bias/ReLU commute with max, so they're applied after pool)
#pragma unroll
    for (int t = 0; t < 4; ++t) {
        const int rt = t >> 1, ct = t & 1;
        const int cc = ct * 16 + nlo;
        v8h p;
#pragma unroll
        for (int j = 0; j < 8; ++j) p[j] = (_Float16)acc[t][j];
        *(v8h*)&pool[wave][rt][cc][hi * 8] = p;
    }
    __syncthreads();

    // Each lane pools one pooled col (pc = lane>>1) x 8 contiguous couts.
    {
        const int pc = lane >> 1;
        const int cobase = (lane & 1) * 8;
        v8h q00 = *(const v8h*)&pool[wave][0][2 * pc][cobase];
        v8h q01 = *(const v8h*)&pool[wave][0][2 * pc + 1][cobase];
        v8h q10 = *(const v8h*)&pool[wave][1][2 * pc][cobase];
        v8h q11 = *(const v8h*)&pool[wave][1][2 * pc + 1][cobase];
        v4f b0 = *(const v4f*)&bias[cbase + cobase];
        v4f b1 = *(const v4f*)&bias[cbase + cobase + 4];
        v8h r;
#pragma unroll
        for (int i = 0; i < 8; ++i) {
            float mx = fmaxf(fmaxf((float)q00[i], (float)q01[i]),
                             fmaxf((float)q10[i], (float)q11[i]));
            const float bi = (i < 4) ? b0[i & 3] : b1[i & 3];
            r[i] = (_Float16)fmaxf(mx + bi, 0.f);
        }
        const int px = (sc0 + wc0) / 2 + pc;
        if (px < WPD)
            *(v8h*)&out[((size_t)(n * HP + py) * WPD + px) * COUT + cbase + cobase] = r;
    }
}

// ---------------------------------------------------------------------------
// x (16,1,100,4000) f32  ->  xs (128,100,500) f16  (segment split, NHWC C=1)
// ---------------------------------------------------------------------------
__global__ void segment_kernel(const float* __restrict__ x,
                               _Float16* __restrict__ xs) {
    const size_t i = (size_t)blockIdx.x * 256 + threadIdx.x;
    if (i >= 6400000ull) return;
    const int r = (int)(i % 500);
    const size_t t = i / 500;
    const int h = (int)(t % 100);
    const int s = (int)(t / 100);       // s = b*8 + l
    const int b = s >> 3, l = s & 7;
    xs[i] = (_Float16)x[((size_t)b * 100 + h) * 4000 + (size_t)l * 500 + r];
}

// ---------------------------------------------------------------------------
// Fold BN into f16 weights packed [Cout][KP] with K = tap*CIN + c (KP mult 32)
// bias' = (b - mean)*g*rsqrt(v+eps) + beta
// ---------------------------------------------------------------------------
template <int CIN, int COUT>
__global__ void pack_w_kernel(const float* __restrict__ w,
                              const float* __restrict__ b,
                              const float* __restrict__ g,
                              const float* __restrict__ be,
                              const float* __restrict__ m,
                              const float* __restrict__ v,
                              _Float16* __restrict__ wp,
                              float* __restrict__ bias2) {
    constexpr int KTOT = 25 * CIN;
    constexpr int KP   = ((KTOT + 31) / 32) * 32;
    const int i = blockIdx.x * 256 + threadIdx.x;
    if (i >= COUT * KP) return;
    const int co = i / KP;
    const int k  = i % KP;
    const float inv = g[co] * rsqrtf(v[co] + 1e-5f);
    _Float16 val = (_Float16)0;
    if (k < KTOT) {
        const int tap = k / CIN, c = k % CIN;
        const int dy = tap / 5, dx = tap % 5;
        val = (_Float16)(w[((size_t)(co * CIN + c) * 5 + dy) * 5 + dx] * inv);
    }
    wp[i] = val;
    if (k == 0) bias2[co] = (b[co] - m[co]) * inv + be[co];
}

// ---------------------------------------------------------------------------
// fc1: (128, 5952) @ fw1.T + fb1 -> (128, 32).  h4 is NHWC (6,31,32) f16; the
// reference flattens NCHW, so fw1 index is k = c*186 + y*31 + x.
// Thread (o = t&31, q = t>>5): channels [q*8, q*8+8) via v8h loads, no divs.
// ---------------------------------------------------------------------------
__global__ void __launch_bounds__(128)
fc1_kernel(const _Float16* __restrict__ h4, const float* __restrict__ fw1,
           const float* __restrict__ fb1, float* __restrict__ out) {
    const int s = blockIdx.x;           // segment-sample 0..127
    const int t = threadIdx.x;
    const int o = t & 31;               // output unit
    const int q = t >> 5;               // channel-slice 0..3
    float acc = 0.f;
    const float* wbase = fw1 + (size_t)o * 5952;
    for (int y = 0; y < 6; ++y) {
        for (int x2 = 0; x2 < 31; ++x2) {
            const v8h hv =
                *(const v8h*)&h4[((size_t)(s * 6 + y) * 31 + x2) * 32 + q * 8];
            const float* wr = wbase + y * 31 + x2;
#pragma unroll
            for (int i = 0; i < 8; ++i)
                acc += (float)hv[i] * wr[(q * 8 + i) * 186];
        }
    }
    __shared__ float red[128];
    red[t] = acc;
    __syncthreads();
    if (q == 0)
        out[s * 32 + o] = red[o] + red[32 + o] + red[64 + o] + red[96 + o] + fb1[o];
}

// fc2 + mean over the 8 segments per sample -> (16,2) f32
__global__ void fc2_mean_kernel(const float* __restrict__ fc1,
                                const float* __restrict__ fw2,
                                const float* __restrict__ fb2,
                                float* __restrict__ out) {
    const int t = threadIdx.x;
    if (t >= 32) return;
    const int b = t >> 1, o = t & 1;
    float s = 0.f;
    for (int l = 0; l < 8; ++l) {
        const float* hh = &fc1[(size_t)(b * 8 + l) * 32];
        float d = fb2[o];
#pragma unroll
        for (int c = 0; c < 32; ++c) d += hh[c] * fw2[o * 32 + c];
        s += d;
    }
    out[b * 2 + o] = s * 0.125f;
}

// ---------------------------------------------------------------------------
extern "C" void kernel_launch(void* const* d_in, const int* in_sizes, int n_in,
                              void* d_out, int out_size, void* d_ws,
                              size_t ws_size, hipStream_t stream) {
    (void)in_sizes; (void)n_in; (void)out_size; (void)ws_size;
    const float* x = (const float*)d_in[0];
    const float *w1 = (const float*)d_in[1],  *b1 = (const float*)d_in[2],
                *g1 = (const float*)d_in[3],  *be1 = (const float*)d_in[4],
                *m1 = (const float*)d_in[5],  *v1 = (const float*)d_in[6];
    const float *w2 = (const float*)d_in[7],  *b2 = (const float*)d_in[8],
                *g2 = (const float*)d_in[9],  *be2 = (const float*)d_in[10],
                *m2 = (const float*)d_in[11], *v2 = (const float*)d_in[12];
    const float *w3 = (const float*)d_in[13], *b3 = (const float*)d_in[14],
                *g3 = (const float*)d_in[15], *be3 = (const float*)d_in[16],
                *m3 = (const float*)d_in[17], *v3 = (const float*)d_in[18];
    const float *w4 = (const float*)d_in[19], *b4 = (const float*)d_in[20],
                *g4 = (const float*)d_in[21], *be4 = (const float*)d_in[22],
                *m4 = (const float*)d_in[23], *v4 = (const float*)d_in[24];
    const float *fw1 = (const float*)d_in[25], *fb1 = (const float*)d_in[26];
    const float *fw2 = (const float*)d_in[27], *fb2 = (const float*)d_in[28];

    // ---- workspace layout (two ping-pong activation buffers + weights) ----
    char* ws = (char*)d_ws;
    size_t off = 0;
    auto alloc = [&](size_t bytes) -> char* {
        off = (off + 255) & ~(size_t)255;
        char* p = ws + off;
        off += bytes;
        return p;
    };
    // bufA: xs (6.4M h), h2 (12.8M h), h4 (0.76M h) -> 12.8M halves
    // bufB: h1 (25.6M h), h3 (6.1M h)               -> 25.6M halves
    _Float16* bufA = (_Float16*)alloc(12800000ull * 2);
    _Float16* bufB = (_Float16*)alloc(25600000ull * 2);
    _Float16* wp1 = (_Float16*)alloc(16 * 32 * 2);
    _Float16* wp2 = (_Float16*)alloc(32 * 416 * 2);
    _Float16* wp3 = (_Float16*)alloc(64 * 800 * 2);
    _Float16* wp4 = (_Float16*)alloc(32 * 1600 * 2);
    float* bb1 = (float*)alloc(16 * 4);
    float* bb2 = (float*)alloc(32 * 4);
    float* bb3 = (float*)alloc(64 * 4);
    float* bb4 = (float*)alloc(32 * 4);
    float* fc1o = (float*)alloc(128 * 32 * 4);

    _Float16* xs = bufA;   // (128,100,500,1)
    _Float16* h1 = bufB;   // (128, 50,250,16)
    _Float16* h2 = bufA;   // (128, 25,125,32)
    _Float16* h3 = bufB;   // (128, 12, 62,64)
    _Float16* h4 = bufA;   // (128,  6, 31,32)

    // ---- prep ----
    segment_kernel<<<25000, 256, 0, stream>>>(x, xs);
    pack_w_kernel<1, 16><<<2, 256, 0, stream>>>(w1, b1, g1, be1, m1, v1, wp1, bb1);
    pack_w_kernel<16, 32><<<52, 256, 0, stream>>>(w2, b2, g2, be2, m2, v2, wp2, bb2);
    pack_w_kernel<32, 64><<<200, 256, 0, stream>>>(w3, b3, g3, be3, m3, v3, wp3, bb3);
    pack_w_kernel<64, 32><<<200, 256, 0, stream>>>(w4, b4, g4, be4, m4, v4, wp4, bb4);

    // ---- fused conv blocks (WMMA implicit GEMM) ----
    // grid = n(128) * HP * XB   (XB = ceil(WP / (16*SP)))
    conv_bnrp_kernel<1, 16, 100, 500><<<128 * 50 * 4, 128, 0, stream>>>(xs, wp1, bb1, h1);
    conv_bnrp_kernel<16, 32, 50, 250><<<128 * 25 * 4, 128, 0, stream>>>(h1, wp2, bb2, h2);
    conv_bnrp_kernel<32, 64, 25, 125><<<128 * 12 * 4, 128, 0, stream>>>(h2, wp3, bb3, h3);
    conv_bnrp_kernel<64, 32, 12, 62><<<128 * 6 * 1, 128, 0, stream>>>(h3, wp4, bb4, h4);

    // ---- head ----
    fc1_kernel<<<128, 128, 0, stream>>>(h4, fw1, fb1, fc1o);
    fc2_mean_kernel<<<1, 32, 0, stream>>>(fc1o, fw2, fb2, (float*)d_out);
}
